// LSTM_AD_13657996002074
// MI455X (gfx1250) — matrix operand order repaired and checked by reading, not compiled, not run on hardware
//
#include <hip/hip_runtime.h>
#include <hip/hip_bf16.h>
#include <stdint.h>

typedef __attribute__((ext_vector_type(16))) _Float16 v16h;
typedef __attribute__((ext_vector_type(8)))  _Float16 v8h;
typedef __attribute__((ext_vector_type(4)))  _Float16 v4h;
typedef __attribute__((ext_vector_type(8)))  float    v8f;

#define T_SEQ 65536
#define FEAT  512
#define HID   64
#define GATES 256   // 4*HID
#define DEPTH 8     // async xg ring depth (rows in flight)

// ---- CDNA5 async memory helpers (not covered by builtins) ------------------
__device__ __forceinline__ void async_lds_b64(unsigned lds_off, const void* gaddr) {
  // GLOBAL_LOAD_ASYNC_TO_LDS_B64: DMA 8B/lane global -> LDS, tracked by ASYNCcnt
  asm volatile("global_load_async_to_lds_b64 %0, %1, off"
               :: "v"(lds_off), "v"(gaddr) : "memory");
}
__device__ __forceinline__ void wait_async_le7() {
  asm volatile("s_wait_asynccnt 0x7" ::: "memory");
}

__device__ __forceinline__ float fast_sigmoid(float x) {
  return 1.0f / (1.0f + __expf(-x));
}
__device__ __forceinline__ float fast_tanh(float x) {
  float xc = fminf(fmaxf(x, -15.f), 15.f);     // avoid inf/inf
  float e  = __expf(2.0f * xc);
  return (e - 1.0f) / (e + 1.0f);
}

// ---------------------------------------------------------------------------
// Kernel 0: convert W_ih [256,512] f32 -> f16 row-major (one-time prep).
// ---------------------------------------------------------------------------
__global__ void wih_to_f16(const float* __restrict__ W, _Float16* __restrict__ O, int n) {
  int i = blockIdx.x * blockDim.x + threadIdx.x;
  if (i < n) O[i] = (_Float16)W[i];
}

// ---------------------------------------------------------------------------
// Kernel 1: xg[t,g] = sum_k x[t,k]*W_ih[g,k] + b_ih[g] + b_hh[g]
// Memory-bound (~192MB @ 23.3TB/s ~= 8us floor). 16-row M-tile staged in LDS
// as f16 once; 8 waves x 2 N-tiles; 16 K-chunks of v_wmma_f32_16x16x32_f16.
// ---------------------------------------------------------------------------
__global__ __launch_bounds__(256) void gemm_xg(
    const float*    __restrict__ x,
    const _Float16* __restrict__ W16,
    const float*    __restrict__ b_ih,
    const float*    __restrict__ b_hh,
    float*          __restrict__ xg)
{
  __shared__ _Float16 xt[16 * FEAT];           // 16KB f16 x-tile

  const int tid  = threadIdx.x;
  const int lane = tid & 31;
  const int wave = tid >> 5;                   // 8 waves
  const int m0   = blockIdx.x * 16;
  const int hiHalf = (lane & 16) ? 1 : 0;
  const int ln16 = lane & 15;

  // stage x tile (16x512 f32, contiguous 32KB) into LDS as f16
  const float4* xs = (const float4*)(x + (size_t)m0 * FEAT);
  #pragma unroll
  for (int i = 0; i < 8; ++i) {
    int e4 = tid + 256 * i;
    float4 v = xs[e4];
    v4h h;
    h[0] = (_Float16)v.x; h[1] = (_Float16)v.y;
    h[2] = (_Float16)v.z; h[3] = (_Float16)v.w;
    *(v4h*)&xt[e4 * 4] = h;
  }
  __syncthreads();

  const int tb0 = (2 * wave) * 16;             // this wave's two N-tiles
  const int tb1 = tb0 + 16;
  const int m   = ln16;
  v8f acc0 = {}; v8f acc1 = {};

  #pragma unroll 4
  for (int c = 0; c < 16; ++c) {               // K chunks of 32
    int kb = c * 32 + hiHalf * 16;             // A: 16 contiguous K halves
    v8h alo = *(const v8h*)&xt[m * FEAT + kb];
    v8h ahi = *(const v8h*)&xt[m * FEAT + kb + 8];
    v16h a = __builtin_shufflevector(alo, ahi, 0,1,2,3,4,5,6,7,8,9,10,11,12,13,14,15);

    int ks = c * 32 + hiHalf * 16;             // B: 16 contiguous K halves
    const v8h* w0 = (const v8h*)(W16 + (size_t)(tb0 + ln16) * FEAT + ks);
    const v8h* w1 = (const v8h*)(W16 + (size_t)(tb1 + ln16) * FEAT + ks);
    v16h b0 = __builtin_shufflevector(w0[0], w0[1], 0,1,2,3,4,5,6,7,8,9,10,11,12,13,14,15);
    v16h b1 = __builtin_shufflevector(w1[0], w1[1], 0,1,2,3,4,5,6,7,8,9,10,11,12,13,14,15);

    acc0 = __builtin_amdgcn_wmma_f32_16x16x32_f16(false, a, false, b0, (short)0, acc0, false, false);
    acc1 = __builtin_amdgcn_wmma_f32_16x16x32_f16(false, a, false, b1, (short)0, acc1, false, false);
  }

  int g0 = tb0 + ln16, g1 = tb1 + ln16;
  float bb0 = b_ih[g0] + b_hh[g0];
  float bb1 = b_ih[g1] + b_hh[g1];
  #pragma unroll
  for (int v = 0; v < 8; ++v) {
    int row = v + hiHalf * 8;
    xg[(size_t)(m0 + row) * GATES + g0] = acc0[v] + bb0;
    xg[(size_t)(m0 + row) * GATES + g1] = acc1[v] + bb1;
  }
}

// ---------------------------------------------------------------------------
// Kernel 2: sequential LSTM scan + final FC. One persistent block, 4 waves.
// Wave w owns gates [64w,64w+64): W_hh resident in VGPRs as f16 B fragments,
// h replicated across A rows -> 8 WMMAs/step/wave.
//  - ONE barrier/step: gate exchange is ping-pong double-buffered; the c/h
//    update is computed redundantly by every wave into a per-wave private
//    f16 h buffer, so the h-write -> A-read dependency is intra-wave (DScnt).
//  - xg rows stream through an 8-deep LDS ring via GLOBAL_LOAD_ASYNC_TO_LDS,
//    each wave DMAs exactly the 256B segment it reads (per-wave ASYNCcnt is
//    in-order, so s_wait_asynccnt 7 guarantees the oldest row landed).
// ---------------------------------------------------------------------------
__global__ __launch_bounds__(128, 1) void lstm_scan(
    const float* __restrict__ xg,
    const float* __restrict__ h0,
    const float* __restrict__ c0,
    const float* __restrict__ W_hh,
    const float* __restrict__ W_fc,
    const float* __restrict__ b_fc,
    float*       __restrict__ out)
{
  __shared__ _Float16 h16w[4][HID];         // per-wave private h (f16)
  __shared__ float    gts[2][4][HID];       // ping-pong gate exchange
  __shared__ float    xring[DEPTH][GATES];  // async-staged xg rows (8KB)
  __shared__ float    h32[HID];

  const int tid    = threadIdx.x;
  const int lane   = tid & 31;
  const int wave   = tid >> 5;              // 0..3 -> gate group i/f/g/o
  const int ln16   = lane & 15;
  const int hiHalf = (lane & 16) ? 1 : 0;

  // ---- preload W_hh as f16 B fragments (once) ----
  v16h Bt[4][2];
  #pragma unroll
  for (int t = 0; t < 4; ++t) {
    #pragma unroll
    for (int c = 0; c < 2; ++c) {
      int g  = 64 * wave + 16 * t + ln16;
      int ks = c * 32 + hiHalf * 16;
      const float4* wp = (const float4*)(W_hh + (size_t)g * HID + ks);
      v16h b;
      #pragma unroll
      for (int q = 0; q < 4; ++q) {
        float4 f = wp[q];
        b[4*q+0] = (_Float16)f.x; b[4*q+1] = (_Float16)f.y;
        b[4*q+2] = (_Float16)f.z; b[4*q+3] = (_Float16)f.w;
      }
      Bt[t][c] = b;
    }
  }

  // ---- replicated state: every wave carries full c (2 units/lane) ----
  float c_a = c0[lane],      c_b = c0[lane + 32];
  float h_a = h0[lane],      h_b = h0[lane + 32];
  _Float16* myh = h16w[wave];
  myh[lane]      = (_Float16)h_a;
  myh[lane + 32] = (_Float16)h_b;           // intra-wave buffer: no barrier

  // ---- async ring warm-up: rows 0..DEPTH-1, 8B per lane per row ----
  const unsigned ldsSeg = (unsigned)(uintptr_t)&xring[0][64 * wave + lane * 2];
  const float*   gSeg   = xg + 64 * wave + lane * 2;
  #pragma unroll
  for (int d = 0; d < DEPTH; ++d)
    async_lds_b64(ldsSeg + (unsigned)(d * GATES * 4), gSeg + (size_t)d * GATES);

  const int kb0 = hiHalf * 16;

  #pragma unroll 1
  for (int step = 0; step < T_SEQ; ++step) {
    const int slot = step & (DEPTH - 1);

    // oldest in-flight row (this step's) has landed when <=7 outstanding
    wait_async_le7();
    const float* xr = &xring[slot][64 * wave + ln16];
    float xv0 = xr[0], xv1 = xr[16], xv2 = xr[32], xv3 = xr[48];
    asm volatile("" ::: "memory");          // keep ring reads before refill
    int nrow = step + DEPTH; if (nrow > T_SEQ - 1) nrow = T_SEQ - 1;
    async_lds_b64(ldsSeg + (unsigned)(slot * GATES * 4), gSeg + (size_t)nrow * GATES);

    // A fragments (replicated h) from this wave's private buffer
    v8h a0lo = *(const v8h*)&myh[kb0];
    v8h a0hi = *(const v8h*)&myh[kb0 + 8];
    v8h a1lo = *(const v8h*)&myh[kb0 + 32];
    v8h a1hi = *(const v8h*)&myh[kb0 + 40];
    v16h a0 = __builtin_shufflevector(a0lo, a0hi, 0,1,2,3,4,5,6,7,8,9,10,11,12,13,14,15);
    v16h a1 = __builtin_shufflevector(a1lo, a1hi, 0,1,2,3,4,5,6,7,8,9,10,11,12,13,14,15);

    // 4 tiles x 2 WMMA -> 64 gate preactivations for this wave
    float p[4];
    #pragma unroll
    for (int t = 0; t < 4; ++t) {
      v8f acc = {};
      acc = __builtin_amdgcn_wmma_f32_16x16x32_f16(false, a0, false, Bt[t][0], (short)0, acc, false, false);
      acc = __builtin_amdgcn_wmma_f32_16x16x32_f16(false, a1, false, Bt[t][1], (short)0, acc, false, false);
      p[t] = acc[0];                        // row 0 (lanes 0..15)
    }
    p[0] += xv0; p[1] += xv1; p[2] += xv2; p[3] += xv3;

    float* gw = gts[step & 1][wave];
    #pragma unroll
    for (int t = 0; t < 4; ++t) {
      float act = (wave == 2) ? fast_tanh(p[t]) : fast_sigmoid(p[t]);
      if (lane < 16) gw[16 * t + ln16] = act;
    }
    __syncthreads();                        // the ONLY barrier per step

    // redundant c/h update in every wave (2 hidden units per lane)
    const float* gp = &gts[step & 1][0][0];
    float iv1 = gp[lane],       fv1 = gp[64 + lane];
    float gv1 = gp[128 + lane], ov1 = gp[192 + lane];
    float iv2 = gp[lane + 32],       fv2 = gp[96 + lane];       // 64+32+lane
    float gv2 = gp[160 + lane],      ov2 = gp[224 + lane];
    c_a = fv1 * c_a + iv1 * gv1;  h_a = ov1 * fast_tanh(c_a);
    c_b = fv2 * c_b + iv2 * gv2;  h_b = ov2 * fast_tanh(c_b);
    myh[lane]      = (_Float16)h_a;
    myh[lane + 32] = (_Float16)h_b;         // intra-wave: DScnt only
  }

  // ---- final: out = sigmoid(W_fc @ h_last + b_fc) ----
  if (wave == 0) { h32[lane] = h_a; h32[lane + 32] = h_b; }
  __syncthreads();
  for (int r = tid; r < FEAT; r += 128) {
    float acc = b_fc[r];
    const float* wr = W_fc + (size_t)r * HID;
    #pragma unroll 8
    for (int k = 0; k < HID; ++k) acc += wr[k] * h32[k];
    out[r] = fast_sigmoid(acc);
  }
}

// ---------------------------------------------------------------------------
extern "C" void kernel_launch(void* const* d_in, const int* in_sizes, int n_in,
                              void* d_out, int out_size, void* d_ws, size_t ws_size,
                              hipStream_t stream) {
  const float* x    = (const float*)d_in[0];
  const float* h0   = (const float*)d_in[1];
  const float* c0   = (const float*)d_in[2];
  const float* W_ih = (const float*)d_in[3];
  const float* W_hh = (const float*)d_in[4];
  const float* b_ih = (const float*)d_in[5];
  const float* b_hh = (const float*)d_in[6];
  const float* W_fc = (const float*)d_in[7];
  const float* b_fc = (const float*)d_in[8];

  // workspace: xg [T,256] f32 (64MB) then W_ih in f16 (256KB)
  float*    xg  = (float*)d_ws;
  _Float16* w16 = (_Float16*)((char*)d_ws + (size_t)T_SEQ * GATES * sizeof(float));

  const int nW = GATES * FEAT;
  wih_to_f16<<<(nW + 255) / 256, 256, 0, stream>>>(W_ih, w16, nW);
  gemm_xg<<<T_SEQ / 16, 256, 0, stream>>>(x, w16, b_ih, b_hh, xg);
  lstm_scan<<<1, 128, 0, stream>>>(xg, h0, c0, W_hh, W_fc, b_fc, (float*)d_out);
}
